// Diffpool_68934225101287
// MI455X (gfx1250) — matrix operand order repaired and checked by reading, not compiled, not run on hardware
//
#include <hip/hip_runtime.h>
#include <math.h>

#define GB 128   // graphs
#define GN 512   // nodes per graph
#define GD 128   // features
#define GC 64    // clusters
#define EPSF 1e-15f

typedef __attribute__((ext_vector_type(16))) _Float16 v16h;
typedef __attribute__((ext_vector_type(8)))  float    v8f;

// ---------- WMMA fragment loaders (A and B both use contiguous k = 16*half + s) ----------

// A (16x32 f16): row-major source [rows x >=k0+32], optional per-row scale 1/max(deg,1)
__device__ inline v16h load_a_frag(const float* __restrict__ base, int row0, int k0, int ld,
                                   const float* __restrict__ deg) {
  const int lane = threadIdx.x & 31;
  const int half = lane >> 4, r = lane & 15;
  const float* rp = base + (size_t)(row0 + r) * ld + k0;
  float sc = 1.0f;
  if (deg) sc = 1.0f / fmaxf(deg[row0 + r], 1.0f);
  v16h a;
#pragma unroll
  for (int v = 0; v < 8; ++v) {
    const int kk = 2 * v + (half << 4);
    a[2 * v]     = (_Float16)(rp[kk] * sc);
    a[2 * v + 1] = (_Float16)(rp[kk + 1] * sc);
  }
  return a;
}

// A (16x32 f16) from a transposed source: A[m,k] = base[k*ld + m]  (A = S^T)
__device__ inline v16h load_at_frag(const float* __restrict__ base, int row0, int k0, int ld) {
  const int lane = threadIdx.x & 31;
  const int half = lane >> 4, r = lane & 15;
  const float* cp = base + (size_t)k0 * ld + row0 + r;
  v16h a;
#pragma unroll
  for (int v = 0; v < 8; ++v) {
    const int kk = 2 * v + (half << 4);
    a[2 * v]     = (_Float16)cp[(size_t)kk * ld];
    a[2 * v + 1] = (_Float16)cp[(size_t)(kk + 1) * ld];
  }
  return a;
}

// B (32x16 f16): row-major source B[k,j] = base[k*ld + col0 + j]
__device__ inline v16h load_b_frag(const float* __restrict__ base, int k0, int col0, int ld) {
  const int lane = threadIdx.x & 31;
  const int half = lane >> 4, j = lane & 15;
  const float* cp = base + (size_t)(k0 + half * 16) * ld + col0 + j;
  v16h bfr;
#pragma unroll
  for (int v = 0; v < 8; ++v) {
    bfr[2 * v]     = (_Float16)cp[(size_t)(2 * v) * ld];
    bfr[2 * v + 1] = (_Float16)cp[(size_t)(2 * v + 1) * ld];
  }
  return bfr;
}

// B (32x16 f16) from transposed source: B[k,j] = base[(col0+j)*ld + k]  (B = W^T, contiguous reads)
__device__ inline v16h load_bt_frag(const float* __restrict__ base, int k0, int col0, int ld) {
  const int lane = threadIdx.x & 31;
  const int half = lane >> 4, j = lane & 15;
  const float* rp = base + (size_t)(col0 + j) * ld + k0 + half * 16;
  v16h bfr;
#pragma unroll
  for (int v = 0; v < 8; ++v) {
    bfr[2 * v]     = (_Float16)rp[2 * v];
    bfr[2 * v + 1] = (_Float16)rp[2 * v + 1];
  }
  return bfr;
}

// ---------- Pass A over edges: deg, agg = A*X, and ||adj||_F^2 via atomic pre-op values ----------
__global__ void edge_pass_a(const float* __restrict__ x, const int* __restrict__ esrc,
                            const int* __restrict__ edst, int E,
                            float* deg, float* agg, unsigned* cnt, float* adjSq) {
  __shared__ float bsum;
  if (threadIdx.x == 0) bsum = 0.0f;
  __syncthreads();
  const int wave = threadIdx.x >> 5;
  const int lane = threadIdx.x & 31;
  const int eid = blockIdx.x * 8 + wave;
  if (eid < E) {
    const int u = esrc[eid];
    const int v = edst[eid];
    const float* xs = x + (size_t)v * GD;
    float* ag = agg + (size_t)u * GD;
#pragma unroll
    for (int c = lane; c < GD; c += 32) atomicAdd(&ag[c], xs[c]);
    if (lane == 0) {
      atomicAdd(&deg[u], 1.0f);
      // sum_cells k^2 = sum_edges (2*count_before + 1); packed byte counters (4/word)
      const int cell = u * GN + (v & (GN - 1));
      const int sh = (cell & 3) * 8;
      const unsigned old = atomicAdd(&cnt[cell >> 2], 1u << sh);
      atomicAdd(&bsum, 2.0f * (float)((old >> sh) & 0xFFu) + 1.0f);
    }
  }
  __syncthreads();
  if (threadIdx.x == 0) atomicAdd(adjSq, bsum);
}

// ---------- Fused logits GEMM (K=256: [agg/deg | x] @ [Wrel;Wroot]^T + b) + softmax + entropy ----------
__global__ void logits_softmax(const float* __restrict__ agg, const float* __restrict__ x,
                               const float* __restrict__ deg,
                               const float* __restrict__ Wrel, const float* __restrict__ Wroot,
                               const float* __restrict__ brt,
                               float* __restrict__ sOut, float* entAcc) {
  const int gw = blockIdx.x * 8 + (threadIdx.x >> 5);   // wave-uniform
  if (gw >= (GB * GN) / 16) return;
  const int lane = threadIdx.x & 31;
  const int half = lane >> 4, col = lane & 15;
  const int row0 = gw * 16;

  v8f acc[4] = {};
#pragma unroll
  for (int kc = 0; kc < 8; ++kc) {
    const int k0 = kc * 32;
    const bool rel = (k0 < GD);
    const v16h af = rel ? load_a_frag(agg, row0, k0, GD, deg)
                        : load_a_frag(x, row0, k0 - GD, GD, nullptr);
    const float* Wb = rel ? Wrel : Wroot;
    const int kw = rel ? k0 : k0 - GD;
#pragma unroll
    for (int t = 0; t < 4; ++t) {
      const v16h bf = load_bt_frag(Wb, kw, t * 16, GD);
      acc[t] = __builtin_amdgcn_wmma_f32_16x16x32_f16(false, af, false, bf,
                                                      (short)0, acc[t], false, false);
    }
  }
  // bias
#pragma unroll
  for (int t = 0; t < 4; ++t) {
    const float bc = brt[t * 16 + col];
#pragma unroll
    for (int i = 0; i < 8; ++i) acc[t][i] += bc;
  }
  // softmax over C=64: each row lives in 16 lanes (one half) x 4 tiles
  float entLocal = 0.0f;
#pragma unroll
  for (int r = 0; r < 8; ++r) {
    float mx = acc[0][r];
#pragma unroll
    for (int t = 1; t < 4; ++t) mx = fmaxf(mx, acc[t][r]);
#pragma unroll
    for (int m = 1; m < 16; m <<= 1) mx = fmaxf(mx, __shfl_xor(mx, m, 32));
    float e[4], sum = 0.0f;
#pragma unroll
    for (int t = 0; t < 4; ++t) { e[t] = __expf(acc[t][r] - mx); sum += e[t]; }
#pragma unroll
    for (int m = 1; m < 16; m <<= 1) sum += __shfl_xor(sum, m, 32);
    const float inv = 1.0f / sum;
    float* sp = sOut + (size_t)(row0 + half * 8 + r) * GC;
#pragma unroll
    for (int t = 0; t < 4; ++t) {
      const float sv = e[t] * inv;
      sp[t * 16 + col] = sv;
      entLocal -= sv * __logf(sv + EPSF);
    }
  }
#pragma unroll
  for (int m = 1; m < 32; m <<= 1) entLocal += __shfl_xor(entLocal, m, 32);
  if (lane == 0) atomicAdd(entAcc, entLocal);
}

// ---------- Pass B over edges: Y = adj*S and <adj, S S^T> ----------
__global__ void edge_pass_b(const float* __restrict__ sM, const int* __restrict__ esrc,
                            const int* __restrict__ edst, int E,
                            float* Y, float* linkDot) {
  __shared__ float bsum;
  if (threadIdx.x == 0) bsum = 0.0f;
  __syncthreads();
  const int wave = threadIdx.x >> 5, lane = threadIdx.x & 31;
  const int eid = blockIdx.x * 8 + wave;
  float p = 0.0f;
  if (eid < E) {
    const int u = esrc[eid], v = edst[eid];
    const float* su = sM + (size_t)u * GC;
    const float* sv = sM + (size_t)v * GC;
    float* yu = Y + (size_t)u * GC;
#pragma unroll
    for (int c = lane; c < GC; c += 32) {
      const float svv = sv[c];
      p += su[c] * svv;
      atomicAdd(&yu[c], svv);
    }
  }
#pragma unroll
  for (int m = 1; m < 32; m <<= 1) p += __shfl_xor(p, m, 32);
  if (lane == 0) atomicAdd(&bsum, p);
  __syncthreads();
  if (threadIdx.x == 0) atomicAdd(linkDot, bsum);
}

// ---------- Per-graph pooling GEMM: S^T @ [X | Y | S] -> out_x, out_adj, ||S^T S||^2 ----------
__global__ void pool_gemm(const float* __restrict__ sM, const float* __restrict__ x,
                          const float* __restrict__ Y,
                          float* __restrict__ outx, float* __restrict__ outadj, float* pSq) {
  const int b = blockIdx.x;
  const int wave = threadIdx.x >> 5, lane = threadIdx.x & 31;
  const int half = lane >> 4, col = lane & 15;
  const float* sB = sM + (size_t)b * GN * GC;
  const float* xB = x + (size_t)b * GN * GD;
  const float* yB = Y + (size_t)b * GN * GC;
  float pLocal = 0.0f;
  for (int tile = wave; tile < 64; tile += 8) {   // wave-uniform loop
    const int mt = tile >> 4, nt = tile & 15;
    const int row0 = mt * 16;
    const float* bBase; int ldb, colLoc;
    if (nt < 8)       { bBase = xB; ldb = GD; colLoc = nt * 16; }
    else if (nt < 12) { bBase = yB; ldb = GC; colLoc = (nt - 8) * 16; }
    else              { bBase = sB; ldb = GC; colLoc = (nt - 12) * 16; }
    v8f acc = {};
#pragma unroll 4
    for (int k0 = 0; k0 < GN; k0 += 32) {
      const v16h af = load_at_frag(sB, row0, k0, GC);   // A = S^T
      const v16h bf = load_b_frag(bBase, k0, colLoc, ldb);
      acc = __builtin_amdgcn_wmma_f32_16x16x32_f16(false, af, false, bf,
                                                   (short)0, acc, false, false);
    }
    const int crow = row0 + half * 8;
    if (nt < 8) {
#pragma unroll
      for (int r = 0; r < 8; ++r)
        outx[(size_t)b * GC * GD + (size_t)(crow + r) * GD + colLoc + col] = acc[r];
    } else if (nt < 12) {
#pragma unroll
      for (int r = 0; r < 8; ++r)
        outadj[(size_t)b * GC * GC + (size_t)(crow + r) * GC + colLoc + col] = acc[r];
    } else {
#pragma unroll
      for (int r = 0; r < 8; ++r) pLocal += acc[r] * acc[r];   // ||G||_F^2 contribution
    }
  }
#pragma unroll
  for (int m = 1; m < 32; m <<= 1) pLocal += __shfl_xor(pLocal, m, 32);
  if (lane == 0) atomicAdd(pSq, pLocal);
}

// ---------- new_batch + losses ----------
__global__ void finalize(const float* __restrict__ acc, float* __restrict__ out) {
  const int idx = blockIdx.x * blockDim.x + threadIdx.x;
  const int nbOff = GB * GC * GD + GB * GC * GC;
  if (idx < GB * GC) out[nbOff + idx] = (float)(idx >> 6);  // idx / C
  if (idx == 0) {
    const float v = acc[0] - 2.0f * acc[1] + acc[2];        // ||adj - S S^T||^2
    out[nbOff + GB * GC]     = sqrtf(fmaxf(v, 0.0f)) / (float)((size_t)GB * GN * GN);
    out[nbOff + GB * GC + 1] = acc[3] / (float)(GB * GN);
  }
}

extern "C" void kernel_launch(void* const* d_in, const int* in_sizes, int n_in,
                              void* d_out, int out_size, void* d_ws, size_t ws_size,
                              hipStream_t stream) {
  const float* x     = (const float*)d_in[0];
  const int*   eidx  = (const int*)d_in[1];
  const float* Wrel  = (const float*)d_in[3];
  const float* Wroot = (const float*)d_in[4];
  const float* brt   = (const float*)d_in[5];
  const int E = in_sizes[1] / 2;
  const int* esrc = eidx;
  const int* edst = eidx + E;

  // Workspace layout (floats): deg | agg | s | Y | cnt(u32) | 4 scalars   (~96.3 MB)
  float* ws   = (float*)d_ws;
  float* deg  = ws;
  float* agg  = deg + GB * GN;
  float* sbuf = agg + (size_t)GB * GN * GD;
  float* Y    = sbuf + (size_t)GB * GN * GC;
  unsigned* cnt = (unsigned*)(Y + (size_t)GB * GN * GC);
  float* scal = (float*)(cnt + ((size_t)GB * GN * GN) / 4);  // [adjSq, linkDot, pSq, ent]

  // zero deg+agg (contiguous) and Y+cnt+scalars (contiguous); s is fully overwritten
  hipMemsetAsync(deg, 0, sizeof(float) * (size_t)GB * GN * (GD + 1), stream);
  hipMemsetAsync(Y, 0,
                 sizeof(float) * (size_t)GB * GN * GC +
                 sizeof(unsigned) * ((size_t)GB * GN * GN / 4) + 4 * sizeof(float), stream);

  float* outx   = (float*)d_out;
  float* outadj = outx + (size_t)GB * GC * GD;

  const int eblocks = (E + 7) / 8;
  edge_pass_a<<<eblocks, 256, 0, stream>>>(x, esrc, edst, E, deg, agg, cnt, &scal[0]);
  logits_softmax<<<(GB * GN / 16 + 7) / 8, 256, 0, stream>>>(agg, x, deg, Wrel, Wroot, brt,
                                                             sbuf, &scal[3]);
  edge_pass_b<<<eblocks, 256, 0, stream>>>(sbuf, esrc, edst, E, Y, &scal[1]);
  pool_gemm<<<GB, 256, 0, stream>>>(sbuf, x, Y, outx, outadj, &scal[2]);
  finalize<<<(GB * GC + 255) / 256, 256, 0, stream>>>(scal, (float*)d_out);
}
